// PyramidSparseAttention_24927990186526
// MI455X (gfx1250) — compile-verified, matrix operand order checked by reading, not run on hardware
//
#include <hip/hip_runtime.h>
#include <hip/hip_bf16.h>

// ---- problem constants (from reference) ----
#define BB   4
#define NN   2048
#define CN   768
#define NH   12
#define HD   64
#define BH   (BB * NH)        // 48
#define SSEL 1228             // int(2048 * 0.6)
#define SPAD 1248             // 39 * 32, WMMA-K padded

typedef __attribute__((ext_vector_type(16))) _Float16 v16h;
typedef __attribute__((ext_vector_type(8)))  float    v8f;
typedef _Float16 h16;

__device__ __forceinline__ v8f wmma_f16(v16h a, v16h b, v8f c) {
  // D = A(16x32 f16) * B(32x16 f16) + C(16x16 f32)
  return __builtin_amdgcn_wmma_f32_16x16x32_f16(false, a, false, b, (short)0, c,
                                                false, false);
}

// ---------------------------------------------------------------------------
// Kernel 0: one-time fp32 -> f16 conversion (8 elems/thread, vectorized)
// ---------------------------------------------------------------------------
__global__ void cvt_kernel(const float* __restrict__ in, h16* __restrict__ out,
                           int nelem) {
  const int i = (blockIdx.x * blockDim.x + threadIdx.x) * 8;
  if (i + 8 <= nelem) {
    const float4 a = *(const float4*)(in + i);
    const float4 b = *(const float4*)(in + i + 4);
    h16 t[8];
    t[0] = (h16)a.x; t[1] = (h16)a.y; t[2] = (h16)a.z; t[3] = (h16)a.w;
    t[4] = (h16)b.x; t[5] = (h16)b.y; t[6] = (h16)b.z; t[7] = (h16)b.w;
    *(uint4*)(out + i) = *(const uint4*)t;     // 16 B aligned (i % 8 == 0)
  }
}

// ---------------------------------------------------------------------------
// Kernel 1: projection GEMM  Y = x @ W^T + bias -> [B,H,N,hd] f16.
// 32x64 tile per wave: 2 A-frags x 4 B-frags = 8 WMMA / k-step, register
// double-buffered; unroll-2 so the two buffers ping-pong without movs.
// ---------------------------------------------------------------------------
__global__ void proj_kernel(const h16* __restrict__ xh,
                            const h16* __restrict__ Wh,
                            const float* __restrict__ bias,
                            h16* __restrict__ out /* [BH][N][HD] */) {
  const int wave = threadIdx.x >> 5;
  const int lane = threadIdx.x & 31;
  const int tile = blockIdx.x * 8 + wave;      // (8192/32)*(768/64) = 3072
  const int TN = CN / 64;                      // 12 column groups (== heads)
  const int tm = tile / TN;
  const int tn = tile % TN;

  const int ln = lane & 15;
  const int g  = lane >> 4;
  const int koff = g * 16;

  const h16* xa0 = xh + (size_t)(tm * 32 + ln) * CN + koff;
  const h16* xa1 = xa0 + (size_t)16 * CN;
  const h16* wb  = Wh + (size_t)(tn * 64 + ln) * CN + koff;

  v8f acc[2][4];
  {
    v8f z = {};
#pragma unroll
    for (int mi = 0; mi < 2; ++mi)
#pragma unroll
      for (int nt = 0; nt < 4; ++nt) acc[mi][nt] = z;
  }

  // prologue: fragments for k = 0
  v16h a_c[2], b_c[4];
  a_c[0] = *(const v16h*)xa0;
  a_c[1] = *(const v16h*)xa1;
#pragma unroll
  for (int nt = 0; nt < 4; ++nt)
    b_c[nt] = *(const v16h*)(wb + (size_t)(nt * 16) * CN);

#pragma unroll 2
  for (int k0 = 0; k0 < CN; k0 += 32) {
    const int k1 = (k0 + 32 < CN) ? k0 + 32 : 0;   // tail prefetch wraps (unused)
    v16h a_n[2], b_n[4];
    a_n[0] = *(const v16h*)(xa0 + k1);
    a_n[1] = *(const v16h*)(xa1 + k1);
#pragma unroll
    for (int nt = 0; nt < 4; ++nt)
      b_n[nt] = *(const v16h*)(wb + (size_t)(nt * 16) * CN + k1);

#pragma unroll
    for (int nt = 0; nt < 4; ++nt) {
      acc[0][nt] = wmma_f16(a_c[0], b_c[nt], acc[0][nt]);
      acc[1][nt] = wmma_f16(a_c[1], b_c[nt], acc[1][nt]);
    }
    a_c[0] = a_n[0]; a_c[1] = a_n[1];
#pragma unroll
    for (int nt = 0; nt < 4; ++nt) b_c[nt] = b_n[nt];
  }

  const int h     = tn;                        // 64-aligned group == head
  const int bbase = (tm * 32) / NN;
  const int nbase = (tm * 32) % NN;
#pragma unroll
  for (int mi = 0; mi < 2; ++mi) {
#pragma unroll
    for (int nt = 0; nt < 4; ++nt) {
      const int d  = nt * 16 + ln;
      const float bj = bias[tn * 64 + d];
#pragma unroll
      for (int i = 0; i < 8; ++i) {
        const int n = nbase + mi * 16 + i + 8 * g;   // C-layout row
        out[(((size_t)bbase * NH + h) * NN + n) * HD + d] =
            (h16)(acc[mi][nt][i] + bj);
      }
    }
  }
}

// ---------------------------------------------------------------------------
// Kernel 2: Qmean[bh][d] = mean over n of Q[bh][n][d]
// ---------------------------------------------------------------------------
__global__ void qmean_kernel(const h16* __restrict__ Q, float* __restrict__ qmean) {
  const int bh = blockIdx.x;
  const int d  = threadIdx.x;                  // 64 threads
  const h16* p = Q + (size_t)bh * NN * HD + d;
  float s = 0.f;
  for (int n = 0; n < NN; ++n) s += (float)p[(size_t)n * HD];
  qmean[bh * HD + d] = s * (1.0f / NN);
}

// ---------------------------------------------------------------------------
// Kernel 3: column importance + deterministic rank-based top-k.
// av[j] = (Qmean . K[j]) / sqrt(hd); rank(j) = #{i : av[i]>av[j] or tie,i<j}
// idx[rank] = j for rank < SSEL  (matches jax.lax.top_k ordering exactly).
// ---------------------------------------------------------------------------
__global__ void topk_kernel(const h16* __restrict__ K,
                            const float* __restrict__ qmean,
                            int* __restrict__ idx) {
  __shared__ float av[NN];
  __shared__ float qm[HD];
  const int bh = blockIdx.x, t = threadIdx.x;  // 256 threads
  if (t < HD) qm[t] = qmean[bh * HD + t];
  __syncthreads();

  const float inv_scale = 0.125f;              // 1/sqrt(64)
  for (int j = t; j < NN; j += 256) {
    const h16* kr = K + ((size_t)bh * NN + j) * HD;
    float s = 0.f;
    for (int d = 0; d < HD; ++d) s += qm[d] * (float)kr[d];
    av[j] = s * inv_scale;
  }
  __syncthreads();

  for (int j = t; j < NN; j += 256) {
    const float v = av[j];
    int r = 0;
    for (int i = 0; i < NN; ++i) {
      const float u = av[i];
      r += (u > v) || (u == v && i < j);
    }
    if (r < SSEL) idx[bh * SPAD + r] = j;
  }
  for (int r = SSEL + t; r < SPAD; r += 256) idx[bh * SPAD + r] = 0;
}

// ---------------------------------------------------------------------------
// Kernel 4: gather V rows, transposed: Vt[bh][d][s] = V[bh][idx[s]][d]
// (zero for padded s) so the attention's B-operand loads are contiguous.
// ---------------------------------------------------------------------------
__global__ void gathervt_kernel(const h16* __restrict__ V,
                                const int* __restrict__ idx,
                                h16* __restrict__ Vt) {
  const int bh = blockIdx.x, t = threadIdx.x;  // 256 threads
  for (int s = t; s < SPAD; s += 256) {
    const int r  = idx[bh * SPAD + s];
    const bool ok = (s < SSEL);
    const h16* vr = V + ((size_t)bh * NN + r) * HD;
    h16* o = Vt + (size_t)bh * HD * SPAD + s;
    for (int d = 0; d < HD; ++d)
      o[(size_t)d * SPAD] = ok ? vr[d] : (h16)0.0f;
  }
}

// ---------------------------------------------------------------------------
// Kernel 5: sparse attention over the selected keys (flash style, 2 passes).
// One wave per 16-query tile; 4 waves / block.  The idx->K gather is
// software-pipelined 2 deep: idx is fetched 2 chunks ahead, K fragments 1
// chunk ahead, so WMMAs overlap the gather latency.  Chunk loops are
// unroll-2 so buffers ping-pong and pass-2's half-branch is static.
// ---------------------------------------------------------------------------
__global__ void attn_kernel(const h16* __restrict__ Q,
                            const h16* __restrict__ K,
                            const h16* __restrict__ Vt,
                            const int* __restrict__ idx,
                            h16* __restrict__ ctx /* [B][N][C] */) {
  __shared__ __align__(64) _Float16 pt[4][16][32];

  const int wave = threadIdx.x >> 5;
  const int lane = threadIdx.x & 31;
  const int wt   = blockIdx.x * 4 + wave;      // 6144 tiles = 48 bh * 128 mtiles
  const int bh   = wt >> 7;
  const int mt   = wt & 127;
  const int b    = bh / NH, h = bh % NH;
  const int ln   = lane & 15;
  const int g    = lane >> 4;
  const int koffg = g * 16;
  const float inv_scale = 0.125f;

  const int*  idxb = idx + bh * SPAD;
  const h16*  Kb   = K   + (size_t)bh * NN * HD;
  const h16*  Vtb  = Vt  + (size_t)bh * HD * SPAD;
  const h16*  Qrow = Q + ((size_t)bh * NN + mt * 16 + ln) * HD + koffg;

  // Q A-fragments for both K-steps (hd = 64 -> 2 x K=32), reused all passes.
  const v16h a0 = *(const v16h*)(Qrow);
  const v16h a1 = *(const v16h*)(Qrow + 32);

  // ---- pass 1: row max over all selected columns (pipelined gather) ----
  float mrow[8];
#pragma unroll
  for (int i = 0; i < 8; ++i) mrow[i] = -1e30f;

  v16h kb0, kb1;
  int krow_n;
  {
    const int krow_c = idxb[ln];
    const h16* kp = Kb + (size_t)krow_c * HD + koffg;
    kb0 = *(const v16h*)kp;
    kb1 = *(const v16h*)(kp + 32);
    krow_n = idxb[16 + ln];
  }
#pragma unroll 2
  for (int sc = 0; sc < SPAD; sc += 16) {
    const int sc2 = (sc + 32 < SPAD) ? sc + 32 : 0;     // tail wraps (unused)
    const int krow_f = idxb[sc2 + ln];
    const h16* kpn = Kb + (size_t)krow_n * HD + koffg;
    const v16h nb0 = *(const v16h*)kpn;
    const v16h nb1 = *(const v16h*)(kpn + 32);

    v8f c = {};
    c = wmma_f16(a0, kb0, c);
    c = wmma_f16(a1, kb1, c);
    const int col = sc + ln;
    if (col < SSEL) {
#pragma unroll
      for (int i = 0; i < 8; ++i) mrow[i] = fmaxf(mrow[i], c[i] * inv_scale);
    }
    kb0 = nb0; kb1 = nb1; krow_n = krow_f;
  }
#pragma unroll
  for (int off = 1; off < 16; off <<= 1) {
#pragma unroll
    for (int i = 0; i < 8; ++i)
      mrow[i] = fmaxf(mrow[i], __shfl_xor(mrow[i], off, 32));
  }

  // ---- pass 2: exp, rowsum, P @ Vt (same pipeline, 78 chunk stream) ----
  float rsum[8] = {0.f, 0.f, 0.f, 0.f, 0.f, 0.f, 0.f, 0.f};
  v8f acc[4];
  {
    v8f z = {};
#pragma unroll
    for (int q = 0; q < 4; ++q) acc[q] = z;
  }
  {
    const int krow_c = idxb[ln];
    const h16* kp = Kb + (size_t)krow_c * HD + koffg;
    kb0 = *(const v16h*)kp;
    kb1 = *(const v16h*)(kp + 32);
    krow_n = idxb[16 + ln];
  }
#pragma unroll 2
  for (int tc = 0; tc < SPAD / 16; ++tc) {              // 78 chunks, uniform
    const int sc  = tc * 16;
    const int sc2 = (sc + 32 < SPAD) ? sc + 32 : 0;
    const int krow_f = idxb[sc2 + ln];
    const h16* kpn = Kb + (size_t)krow_n * HD + koffg;
    const v16h nb0 = *(const v16h*)kpn;
    const v16h nb1 = *(const v16h*)(kpn + 32);

    v8f c = {};
    c = wmma_f16(a0, kb0, c);
    c = wmma_f16(a1, kb1, c);
    const int col  = sc + ln;
    const bool ok  = (col < SSEL);
    const int half = tc & 1;
#pragma unroll
    for (int i = 0; i < 8; ++i) {
      const float p = ok ? __expf(c[i] * inv_scale - mrow[i]) : 0.0f;
      const h16 ph = (h16)p;
      rsum[i] += (float)ph;                    // sum what we actually multiply
      pt[wave][i + 8 * g][half * 16 + ln] = ph;
    }
    kb0 = nb0; kb1 = nb1; krow_n = krow_f;

    if (half == 1) {                           // static under unroll-2
      __syncthreads();                         // C-layout writes -> A-layout read
      const v16h ap = *(const v16h*)&pt[wave][ln][g * 16];
      __syncthreads();                         // guard next pair's overwrite
      const int s0 = sc - 16;                  // 32-col window start
#pragma unroll
      for (int dt = 0; dt < 4; ++dt) {
        const h16* vp = Vtb + (size_t)(dt * 16 + ln) * SPAD + s0 + koffg;
        const v16h bv = *(const v16h*)vp;
        acc[dt] = wmma_f16(ap, bv, acc[dt]);
      }
    }
  }

#pragma unroll
  for (int off = 1; off < 16; off <<= 1) {
#pragma unroll
    for (int i = 0; i < 8; ++i) rsum[i] += __shfl_xor(rsum[i], off, 32);
  }

  // store ctx in [B,N,C] for the output projection
#pragma unroll
  for (int dt = 0; dt < 4; ++dt) {
    const int d = dt * 16 + ln;
#pragma unroll
    for (int i = 0; i < 8; ++i) {
      const int m = mt * 16 + i + 8 * g;
      const float o = acc[dt][i] / rsum[i];
      ctx[((size_t)b * NN + m) * CN + h * HD + d] = (h16)o;
    }
  }
}

// ---------------------------------------------------------------------------
// Kernel 6: output projection  out = ctx @ Wo^T + bo  (fp32 output).
// Same 32x64 double-buffered tiling as proj_kernel.
// ---------------------------------------------------------------------------
__global__ void outproj_kernel(const h16* __restrict__ ctx,
                               const h16* __restrict__ Woh,
                               const float* __restrict__ bo,
                               float* __restrict__ out) {
  const int wave = threadIdx.x >> 5;
  const int lane = threadIdx.x & 31;
  const int tile = blockIdx.x * 8 + wave;      // 3072 tiles
  const int TN = CN / 64;                      // 12
  const int tm = tile / TN;
  const int tn = tile % TN;
  const int ln = lane & 15;
  const int g  = lane >> 4;
  const int koff = g * 16;

  const h16* xa0 = ctx + (size_t)(tm * 32 + ln) * CN + koff;
  const h16* xa1 = xa0 + (size_t)16 * CN;
  const h16* wb  = Woh + (size_t)(tn * 64 + ln) * CN + koff;

  v8f acc[2][4];
  {
    v8f z = {};
#pragma unroll
    for (int mi = 0; mi < 2; ++mi)
#pragma unroll
      for (int nt = 0; nt < 4; ++nt) acc[mi][nt] = z;
  }

  v16h a_c[2], b_c[4];
  a_c[0] = *(const v16h*)xa0;
  a_c[1] = *(const v16h*)xa1;
#pragma unroll
  for (int nt = 0; nt < 4; ++nt)
    b_c[nt] = *(const v16h*)(wb + (size_t)(nt * 16) * CN);

#pragma unroll 2
  for (int k0 = 0; k0 < CN; k0 += 32) {
    const int k1 = (k0 + 32 < CN) ? k0 + 32 : 0;
    v16h a_n[2], b_n[4];
    a_n[0] = *(const v16h*)(xa0 + k1);
    a_n[1] = *(const v16h*)(xa1 + k1);
#pragma unroll
    for (int nt = 0; nt < 4; ++nt)
      b_n[nt] = *(const v16h*)(wb + (size_t)(nt * 16) * CN + k1);

#pragma unroll
    for (int nt = 0; nt < 4; ++nt) {
      acc[0][nt] = wmma_f16(a_c[0], b_c[nt], acc[0][nt]);
      acc[1][nt] = wmma_f16(a_c[1], b_c[nt], acc[1][nt]);
    }
    a_c[0] = a_n[0]; a_c[1] = a_n[1];
#pragma unroll
    for (int nt = 0; nt < 4; ++nt) b_c[nt] = b_n[nt];
  }

#pragma unroll
  for (int mi = 0; mi < 2; ++mi) {
#pragma unroll
    for (int nt = 0; nt < 4; ++nt) {
      const int j  = tn * 64 + nt * 16 + ln;
      const float bj = bo[j];
#pragma unroll
      for (int i = 0; i < 8; ++i) {
        const int m = tm * 32 + mi * 16 + i + 8 * g;
        out[(size_t)m * CN + j] = acc[mi][nt][i] + bj;
      }
    }
  }
}

// ---------------------------------------------------------------------------
extern "C" void kernel_launch(void* const* d_in, const int* in_sizes, int n_in,
                              void* d_out, int out_size, void* d_ws, size_t ws_size,
                              hipStream_t stream) {
  (void)in_sizes; (void)n_in; (void)out_size; (void)ws_size;
  const float* x  = (const float*)d_in[0];
  const float* Wq = (const float*)d_in[1];
  const float* bq = (const float*)d_in[2];
  const float* Wk = (const float*)d_in[3];
  const float* bk = (const float*)d_in[4];
  const float* Wv = (const float*)d_in[5];
  const float* bv = (const float*)d_in[6];
  const float* Wo = (const float*)d_in[7];
  const float* bo = (const float*)d_in[8];
  float* out = (float*)d_out;

  // workspace carve-up (256 B aligned)
  char* ws = (char*)d_ws;
  size_t off = 0;
  auto carve = [&](size_t bytes) -> char* {
    char* p = ws + off;
    off = (off + bytes + 255) & ~(size_t)255;
    return p;
  };
  const size_t qkvBytes = (size_t)BH * NN * HD * sizeof(h16);   // 12.58 MB
  const int    xElems   = BB * NN * CN;                         // 6,291,456
  const int    wElems   = CN * CN;                              // 589,824
  h16*   xh    = (h16*)carve((size_t)xElems * sizeof(h16));
  h16*   Wqh   = (h16*)carve((size_t)wElems * sizeof(h16));
  h16*   Wkh   = (h16*)carve((size_t)wElems * sizeof(h16));
  h16*   Wvh   = (h16*)carve((size_t)wElems * sizeof(h16));
  h16*   Woh   = (h16*)carve((size_t)wElems * sizeof(h16));
  h16*   Qb    = (h16*)carve(qkvBytes);
  h16*   Kb    = (h16*)carve(qkvBytes);
  h16*   Vb    = (h16*)carve(qkvBytes);
  h16*   ctx   = (h16*)carve((size_t)BB * NN * CN * sizeof(h16));
  h16*   Vt    = (h16*)carve((size_t)BH * HD * SPAD * sizeof(h16));
  float* qmean = (float*)carve((size_t)BH * HD * sizeof(float));
  int*   idx   = (int*)carve((size_t)BH * SPAD * sizeof(int));

  // one-time f16 conversions
  cvt_kernel<<<xElems / 8 / 256, 256, 0, stream>>>(x, xh, xElems);
  cvt_kernel<<<wElems / 8 / 256, 256, 0, stream>>>(Wq, Wqh, wElems);
  cvt_kernel<<<wElems / 8 / 256, 256, 0, stream>>>(Wk, Wkh, wElems);
  cvt_kernel<<<wElems / 8 / 256, 256, 0, stream>>>(Wv, Wvh, wElems);
  cvt_kernel<<<wElems / 8 / 256, 256, 0, stream>>>(Wo, Woh, wElems);

  const int projBlocks = (BB * NN / 32) * (CN / 64) / 8;        // 384
  proj_kernel<<<projBlocks, 256, 0, stream>>>(xh, Wqh, bq, Qb);
  proj_kernel<<<projBlocks, 256, 0, stream>>>(xh, Wkh, bk, Kb);
  proj_kernel<<<projBlocks, 256, 0, stream>>>(xh, Wvh, bv, Vb);

  qmean_kernel<<<BH, HD, 0, stream>>>(Qb, qmean);
  topk_kernel<<<BH, 256, 0, stream>>>(Kb, qmean, idx);
  gathervt_kernel<<<BH, 256, 0, stream>>>(Vb, idx, Vt);

  const int attnBlocks = BH * (NN / 16) / 4;                    // 1536
  attn_kernel<<<attnBlocks, 128, 0, stream>>>(Qb, Kb, Vt, idx, ctx);

  outproj_kernel<<<projBlocks, 256, 0, stream>>>(ctx, Woh, bo, out);
}
